// GraphComparator_74302934220998
// MI455X (gfx1250) — matrix operand order repaired and checked by reading, not compiled, not run on hardware
//
#include <hip/hip_runtime.h>
#include <hip/hip_bf16.h>

typedef __attribute__((ext_vector_type(16))) _Float16 v16h;
typedef __attribute__((ext_vector_type(8)))  _Float16 v8h;
typedef __attribute__((ext_vector_type(8)))  float    v8f;

union hfrag { v16h v; v8h h[2]; };

constexpr int GG = 3;      // graph types
constexpr int BB = 32;     // batch
constexpr int LL = 1024;   // seq len
constexpr int DD = 384;    // hidden
constexpr int GB = GG * BB;                      // 96 folded batches
constexpr int PD = 392;    // padded LDS row stride (halfwords): bank stride 4
constexpr float SCALE = 0.05103103630798288f;    // 1/sqrt(384)

// ---------------------------------------------------------------- kernel 1
// LayerNorm + alignment gather -> f16 a_orig / a_mut rows (zeroed if invalid)
__device__ __forceinline__ void ln_row(const float* __restrict__ x,
                                       const float* __restrict__ w,
                                       const float* __restrict__ bb,
                                       float vf, _Float16* __restrict__ out,
                                       int lane) {
  float v[12];
  float s = 0.f, s2 = 0.f;
#pragma unroll
  for (int k = 0; k < 12; k++) {
    float t = x[lane + k * 32];
    v[k] = t; s += t; s2 += t * t;
  }
#pragma unroll
  for (int m = 1; m < 32; m <<= 1) {
    s  += __shfl_xor(s,  m, 32);
    s2 += __shfl_xor(s2, m, 32);
  }
  float mu  = s  * (1.f / DD);
  float var = s2 * (1.f / DD) - mu * mu;
  float r = rsqrtf(var + 1e-5f);
#pragma unroll
  for (int k = 0; k < 12; k++) {
    int d = lane + k * 32;
    out[d] = (_Float16)(((v[k] - mu) * r * w[d] + bb[d]) * vf);
  }
}

__global__ void k_ln_gather(const float* __restrict__ ho, const float* __restrict__ hm,
                            const int* __restrict__ algn, const int* __restrict__ src,
                            const float* __restrict__ lnw, const float* __restrict__ lnb,
                            _Float16* __restrict__ aO, _Float16* __restrict__ aM) {
  int lane = threadIdx.x & 31;
  int wave = threadIdx.x >> 5;
  long row = (long)blockIdx.x * 8 + wave;       // 0..GB*L-1
  int b = (int)(row / LL);
  int j = (int)(row % LL);
  int a = algn[row];
  int s = src[row];
  bool valid = (a >= 0) && (a < LL);
  int idx = min(max(a, 0), LL - 1);
  int oi = (s == 1) ? idx : j;
  int mi = (s == 1) ? j : idx;
  float vf = valid ? 1.f : 0.f;
  ln_row(ho + ((long)b * LL + oi) * DD, lnw, lnb, vf, aO + row * (long)DD, lane);
  ln_row(hm + ((long)b * LL + mi) * DD, lnw, lnb, vf, aM + row * (long)DD, lane);
}

// ---------------------------------------------------------------- kernel 2
// Pass 1: Z_q = sum_k exp(S[q,k]) (no max needed: |S| <= sqrt(384) for LN rows).
// One wave owns one 16-query tile; key tiles staged in LDS, double-buffered.
__global__ __launch_bounds__(128, 1)
void k_pass1(const _Float16* __restrict__ aO, const _Float16* __restrict__ aM,
             const float* __restrict__ mask, float* __restrict__ rw) {
  __shared__ _Float16 kls[2][16 * PD];
  int t    = threadIdx.x;
  int lane = t & 31;
  int wv   = t >> 5;
  int b    = blockIdx.y;
  int qt   = blockIdx.x * 4 + wv;               // 0..63
  int half = lane >> 4;
  int lr   = lane & 15;
  const long base = (long)b * LL * DD;

  // A fragments: 16 query rows x K=32 per chunk (contiguous K: 16 hwords per lane half)
  hfrag qa[12];
  const _Float16* qrow = aO + base + (long)(qt * 16 + lr) * DD;
#pragma unroll
  for (int c = 0; c < 12; c++) {
    qa[c].h[0] = *(const v8h*)(qrow + c * 32 + half * 16);
    qa[c].h[1] = *(const v8h*)(qrow + c * 32 + half * 16 + 8);
  }

  // cooperative staging: thread -> (row sr, 6 chunks of 8 halfwords)
  int sr = t >> 3;                               // 0..15
  int sc = t & 7;                                // 0..7
  v8h stg[6];
#pragma unroll
  for (int j = 0; j < 6; j++)
    stg[j] = *(const v8h*)(aM + base + (long)sr * DD + (sc + j * 8) * 8);

  float zacc[8];
#pragma unroll
  for (int r = 0; r < 8; r++) zacc[r] = 0.f;

  int cur = 0;
  for (int kt = 0; kt < 64; kt++) {
#pragma unroll
    for (int j = 0; j < 6; j++)
      *(v8h*)(&kls[cur][sr * PD + (sc + j * 8) * 8]) = stg[j];
    __syncthreads();
    if (kt + 1 < 64) {
      const _Float16* ks = aM + base + (long)(kt + 1) * 16 * DD;
#pragma unroll
      for (int j = 0; j < 6; j++)
        stg[j] = *(const v8h*)(ks + (long)sr * DD + (sc + j * 8) * 8);
    }
    v8f s0 = {}, s1 = {};
#pragma unroll
    for (int c = 0; c < 12; c += 2) {
      v16h bf0 = *(const v16h*)(&kls[cur][lr * PD + c * 32 + half * 16]);
      v16h bf1 = *(const v16h*)(&kls[cur][lr * PD + (c + 1) * 32 + half * 16]);
      s0 = __builtin_amdgcn_wmma_f32_16x16x32_f16(false, qa[c].v,     false, bf0,
                                                  (short)0, s0, false, false);
      s1 = __builtin_amdgcn_wmma_f32_16x16x32_f16(false, qa[c + 1].v, false, bf1,
                                                  (short)0, s1, false, false);
    }
#pragma unroll
    for (int r = 0; r < 8; r++)
      zacc[r] += __expf((s0[r] + s1[r]) * SCALE);
    cur ^= 1;
  }

  // one butterfly at the end: Z per row = sum over the 16 lanes of a half
#pragma unroll
  for (int r = 0; r < 8; r++) {
    zacc[r] += __shfl_xor(zacc[r], 1, 32);
    zacc[r] += __shfl_xor(zacc[r], 2, 32);
    zacc[r] += __shfl_xor(zacc[r], 4, 32);
    zacc[r] += __shfl_xor(zacc[r], 8, 32);
  }
  long ro = (long)b * LL + qt * 16 + half * 8;   // this half's 8 rows
  if (lr == 0) {
#pragma unroll
    for (int r = 0; r < 8; r++)
      rw[ro + r] = mask[ro + r] / zacc[r];       // mask_q / Z_q  (Z >= 1024*exp(-19.6) > 0)
  }
}

// ---------------------------------------------------------------- kernel 3
// Pass 2: c[k] = sum_q exp(S[q,k]) * (mask_q/Z_q).  One wave owns one 16-key
// tile (B fragments pinned in VGPRs); query tiles staged in LDS, double-buffered.
__global__ __launch_bounds__(128, 1)
void k_pass2(const _Float16* __restrict__ aO, const _Float16* __restrict__ aM,
             const float* __restrict__ rw, float* __restrict__ cvec) {
  __shared__ _Float16 qls[2][16 * PD];
  int t    = threadIdx.x;
  int lane = t & 31;
  int wv   = t >> 5;
  int b    = blockIdx.y;
  int kt   = blockIdx.x * 4 + wv;               // 0..63
  int half = lane >> 4;
  int lr   = lane & 15;
  const long base = (long)b * LL * DD;

  // B fragments for this key tile (held in registers for all 64 q-tiles)
  v16h kb[12];
  const _Float16* krow = aM + base + (long)(kt * 16 + lr) * DD;
#pragma unroll
  for (int c = 0; c < 12; c++)
    kb[c] = *(const v16h*)(krow + c * 32 + half * 16);

  int sr = t >> 3;
  int sc = t & 7;
  v8h stg[6];
#pragma unroll
  for (int j = 0; j < 6; j++)
    stg[j] = *(const v8h*)(aO + base + (long)sr * DD + (sc + j * 8) * 8);

  float colAcc = 0.f;
  int cur = 0;
  for (int qt = 0; qt < 64; qt++) {
#pragma unroll
    for (int j = 0; j < 6; j++)
      *(v8h*)(&qls[cur][sr * PD + (sc + j * 8) * 8]) = stg[j];
    __syncthreads();
    if (qt + 1 < 64) {
      const _Float16* qs = aO + base + (long)(qt + 1) * 16 * DD;
#pragma unroll
      for (int j = 0; j < 6; j++)
        stg[j] = *(const v8h*)(qs + (long)sr * DD + (sc + j * 8) * 8);
    }
    v8f s0 = {}, s1 = {};
#pragma unroll
    for (int c = 0; c < 12; c += 2) {
      hfrag a0, a1;
      a0.h[0] = *(const v8h*)(&qls[cur][lr * PD + c * 32 + half * 16]);
      a0.h[1] = *(const v8h*)(&qls[cur][lr * PD + c * 32 + half * 16 + 8]);
      a1.h[0] = *(const v8h*)(&qls[cur][lr * PD + (c + 1) * 32 + half * 16]);
      a1.h[1] = *(const v8h*)(&qls[cur][lr * PD + (c + 1) * 32 + half * 16 + 8]);
      s0 = __builtin_amdgcn_wmma_f32_16x16x32_f16(false, a0.v, false, kb[c],
                                                  (short)0, s0, false, false);
      s1 = __builtin_amdgcn_wmma_f32_16x16x32_f16(false, a1.v, false, kb[c + 1],
                                                  (short)0, s1, false, false);
    }
    long ro = (long)b * LL + qt * 16 + half * 8; // rows of this half
    float iv[8];
#pragma unroll
    for (int r = 0; r < 8; r++) iv[r] = rw[ro + r];
#pragma unroll
    for (int r = 0; r < 8; r++)
      colAcc += __expf((s0[r] + s1[r]) * SCALE) * iv[r];
    cur ^= 1;
  }
  colAcc += __shfl_xor(colAcc, 16, 32);          // combine the two row halves
  if (half == 0)
    cvec[(long)b * LL + kt * 16 + lr] = colAcc;
}

// ---------------------------------------------------------------- kernel 4
// diff[b,d] = ( sum_q mask_q aO[q,d] - sum_k c_k aM[k,d] ) / max(sum mask,1e-6)
__global__ void k_colreduce(const _Float16* __restrict__ aO, const _Float16* __restrict__ aM,
                            const float* __restrict__ mask, const float* __restrict__ cvec,
                            float* __restrict__ diff) {
  int b = blockIdx.x;
  int d = threadIdx.x;                           // 384 threads
  long rb = (long)b * LL;
  float aq = 0.f, ak = 0.f, am = 0.f;
  for (int q = 0; q < LL; q++) {
    float mq = mask[rb + q];
    float cq = cvec[rb + q];
    aq += mq * (float)aO[(rb + q) * (long)DD + d];
    ak += cq * (float)aM[(rb + q) * (long)DD + d];
    am += mq;
  }
  diff[(long)b * DD + d] = (aq - ak) / fmaxf(am, 1e-6f);
}

// ---------------------------------------------------------------- kernel 5
// mean over G graph types + 3-layer MLP -> out[32]
__global__ void k_mlp(const float* __restrict__ diff,
                      const float* __restrict__ W1, const float* __restrict__ b1,
                      const float* __restrict__ W2, const float* __restrict__ b2,
                      const float* __restrict__ W3, const float* __restrict__ b3,
                      float* __restrict__ out) {
  __shared__ float f[DD];
  __shared__ float x1[256];
  __shared__ float x2[64];
  int b = blockIdx.x;                            // 0..31
  int t = threadIdx.x;                           // 256 threads
  for (int d = t; d < DD; d += 256)
    f[d] = (diff[(long)(0 * BB + b) * DD + d] +
            diff[(long)(1 * BB + b) * DD + d] +
            diff[(long)(2 * BB + b) * DD + d]) * (1.f / 3.f);
  __syncthreads();
  {
    float acc = b1[t];
    for (int d = 0; d < DD; d++) acc += f[d] * W1[d * 256 + t];
    x1[t] = fmaxf(acc, 0.f);
  }
  __syncthreads();
  if (t < 64) {
    float acc = b2[t];
    for (int i = 0; i < 256; i++) acc += x1[i] * W2[i * 64 + t];
    x2[t] = fmaxf(acc, 0.f);
  }
  __syncthreads();
  if (t == 0) {
    float acc = b3[0];
    for (int i = 0; i < 64; i++) acc += x2[i] * W3[i];
    out[b] = acc;
  }
}

// ---------------------------------------------------------------- launch
extern "C" void kernel_launch(void* const* d_in, const int* in_sizes, int n_in,
                              void* d_out, int out_size, void* d_ws, size_t ws_size,
                              hipStream_t stream) {
  const float* h_orig = (const float*)d_in[0];
  const float* h_mut  = (const float*)d_in[1];
  const int*   algn   = (const int*)  d_in[2];
  const int*   src    = (const int*)  d_in[3];
  const float* mask   = (const float*)d_in[4];
  const float* ln_w   = (const float*)d_in[5];
  const float* ln_b   = (const float*)d_in[6];
  const float* W1     = (const float*)d_in[7];
  const float* b1     = (const float*)d_in[8];
  const float* W2     = (const float*)d_in[9];
  const float* b2     = (const float*)d_in[10];
  const float* W3     = (const float*)d_in[11];
  const float* b3     = (const float*)d_in[12];
  float* out = (float*)d_out;

  const size_t NROWS = (size_t)GB * LL;                 // 98304
  char* w = (char*)d_ws;
  _Float16* aO   = (_Float16*)(w);                      // 75.5 MB
  _Float16* aM   = (_Float16*)(w + 2ull * NROWS * DD);  // 75.5 MB
  float*    rw   = (float*)   (w + 4ull * NROWS * DD);                  // mask/Z
  float*    cvec = (float*)   (w + 4ull * NROWS * DD + 4ull * NROWS);
  float*    diff = (float*)   (w + 4ull * NROWS * DD + 8ull * NROWS);

  k_ln_gather<<<dim3((unsigned)(NROWS / 8)), 256, 0, stream>>>(
      h_orig, h_mut, algn, src, ln_w, ln_b, aO, aM);
  k_pass1<<<dim3(16, GB), 128, 0, stream>>>(aO, aM, mask, rw);
  k_pass2<<<dim3(16, GB), 128, 0, stream>>>(aO, aM, rw, cvec);
  k_colreduce<<<dim3(GB), DD, 0, stream>>>(aO, aM, mask, cvec, diff);
  k_mlp<<<dim3(BB), 256, 0, stream>>>(diff, W1, b1, W2, b2, W3, b3, out);
}
